// kHopAttentionBias_23888608100653
// MI455X (gfx1250) — 8-bit matrix operand order repaired and checked by reading, compiled, not run on hardware
//
#include <hip/hip_runtime.h>

#define NN 1024
#define NEGV -1000000000.0f

typedef int v8i __attribute__((ext_vector_type(8)));

__global__ __launch_bounds__(256) void kZero(uint4* __restrict__ p, int n16) {
    int t = blockIdx.x * 256 + threadIdx.x;
    if (t < n16) p[t] = make_uint4(0u, 0u, 0u, 0u);
}

__global__ __launch_bounds__(256) void kScatter(const int* __restrict__ ei, int E,
                                                unsigned char* __restrict__ A,
                                                unsigned char* __restrict__ At) {
    int e = blockIdx.x * 256 + threadIdx.x;
    if (e >= E) return;
    int s = ei[e];
    int d = ei[E + e];
    A[(size_t)s * NN + d]  = 1;
    At[(size_t)d * NN + s] = 1;
}

__global__ __launch_bounds__(128) void kBMMT(const unsigned char* __restrict__ P,
                                             const unsigned char* __restrict__ Qt,
                                             unsigned char* __restrict__ R) {
    const int lane = threadIdx.x & 31;
    const int wave = threadIdx.x >> 5;
    const int w  = blockIdx.x * 4 + wave;
    const int tm = (w >> 4) * 32;
    const int tn = (w & 15) * 64;
    const int r  = lane & 15;
    const int h  = lane >> 4;

    const unsigned char* prow[2];
    const unsigned char* qrow[4];
#pragma unroll
    for (int i = 0; i < 2; ++i) prow[i] = P  + (size_t)(tm + i * 16 + r) * NN;
#pragma unroll
    for (int j = 0; j < 4; ++j) qrow[j] = Qt + (size_t)(tn + j * 16 + r) * NN;

    v8i acc[2][4] = {};
    for (int kb = 0; kb < NN; kb += 64) {
        v8i a[2], b[4];
#pragma unroll
        for (int i = 0; i < 2; ++i)
#pragma unroll
            for (int v = 0; v < 8; ++v) {
                const int ka = kb + ((v >> 1) << 4) + ((v & 1) << 2) + (h << 3);
                a[i][v] = *(const int*)(prow[i] + ka);
            }
#pragma unroll
        for (int j = 0; j < 4; ++j)
#pragma unroll
            for (int v = 0; v < 8; ++v) {
                const int kq = kb + ((v >> 1) << 4) + ((v & 1) << 2) + (h << 3);
                b[j][v] = *(const int*)(qrow[j] + kq);
            }
#pragma unroll
        for (int i = 0; i < 2; ++i)
#pragma unroll
            for (int j = 0; j < 4; ++j)
                acc[i][j] = __builtin_amdgcn_wmma_i32_16x16x64_iu8(
                    false, a[i], false, b[j], acc[i][j], false, false);
    }

#pragma unroll
    for (int i = 0; i < 2; ++i)
#pragma unroll
        for (int j = 0; j < 4; ++j)
#pragma unroll
            for (int v = 0; v < 8; ++v)
                R[(size_t)(tm + i * 16 + v + (h << 3)) * NN + tn + j * 16 + r] =
                    (acc[i][j][v] > 0) ? 1 : 0;
}

__global__ __launch_bounds__(256) void kBias(const unsigned char* __restrict__ At,
                                             const unsigned char* __restrict__ A2t,
                                             const unsigned char* __restrict__ A3t,
                                             const float* __restrict__ emb,
                                             float* __restrict__ out) {
    __shared__ float semb[96];
    if (threadIdx.x < 96) semb[threadIdx.x] = emb[threadIdx.x];
    __syncthreads();
    int t  = blockIdx.x * 256 + threadIdx.x;
    int a  = t >> 8;
    int b4 = (t & 255) << 2;
    size_t idx = (size_t)a * NN + b4;
    unsigned int h0 = *(const unsigned int*)(At  + idx);
    unsigned int h1 = *(const unsigned int*)(A2t + idx);
    unsigned int h2 = *(const unsigned int*)(A3t + idx);
    int  lab[4];
    bool val[4];
#pragma unroll
    for (int j = 0; j < 4; ++j) {
        unsigned int b0 = (h0 >> (8 * j)) & 0xFFu;
        unsigned int b1 = (h1 >> (8 * j)) & 0xFFu;
        unsigned int b2 = (h2 >> (8 * j)) & 0xFFu;
        val[j] = (b0 | b1 | b2) != 0u;
        lab[j] = b0 ? 0 : (b1 ? 1 : 2);
    }
#pragma unroll
    for (int lh = 0; lh < 32; ++lh) {
        const int l = lh >> 3, hh = lh & 7;
        float4 o;
        o.x = val[0] ? semb[l * 24 + lab[0] * 8 + hh] : NEGV;
        o.y = val[1] ? semb[l * 24 + lab[1] * 8 + hh] : NEGV;
        o.z = val[2] ? semb[l * 24 + lab[2] * 8 + hh] : NEGV;
        o.w = val[3] ? semb[l * 24 + lab[3] * 8 + hh] : NEGV;
        *(float4*)(out + ((size_t)lh << 20) + idx) = o;
    }
}

extern "C" void kernel_launch(void* const* d_in, const int* in_sizes, int n_in,
                              void* d_out, int out_size, void* d_ws, size_t ws_size,
                              hipStream_t stream) {
    (void)n_in; (void)out_size; (void)ws_size;
    const int*   ei  = (const int*)d_in[0];
    const float* emb = (const float*)d_in[1];
    const int E = in_sizes[0] / 2;

    unsigned char* ws  = (unsigned char*)d_ws;
    unsigned char* A   = ws;
    unsigned char* At  = ws + (size_t)(1u << 20);
    unsigned char* A2  = ws + (size_t)(2u << 20);
    unsigned char* A2t = ws + (size_t)(3u << 20);
    unsigned char* A3t = ws + (size_t)(4u << 20);

    const int n16 = (2 * 1024 * 1024) / 16;
    kZero<<<dim3((n16 + 255) / 256), 256, 0, stream>>>((uint4*)A, n16);

    kScatter<<<dim3((E + 255) / 256), 256, 0, stream>>>(ei, E, A, At);

    kBMMT<<<dim3(128), 128, 0, stream>>>(A,  At, A2);
    kBMMT<<<dim3(128), 128, 0, stream>>>(At, A,  A2t);
    kBMMT<<<dim3(128), 128, 0, stream>>>(At, A2, A3t);

    kBias<<<dim3(1024), 256, 0, stream>>>(At, A2t, A3t, emb, (float*)d_out);
}
